// Seq2SeqMulti_2284922602268
// MI455X (gfx1250) — matrix operand order repaired and checked by reading, not compiled, not run on hardware
//
#include <hip/hip_runtime.h>
#include <hip/hip_bf16.h>

// ---------------- constants (match reference) ----------------
#define EIDN  30000
#define EMB   256
#define HIDN  512
#define INDIM 3
#define SRCT  128
#define TRGT  50
#define BSZ   64
#define G3H   1536          // 3*HID
#define XPAD  288           // EMB+1 padded up to a multiple of 32
#define YK    768           // EMB + HID (rate head K)

typedef __bf16 bf16_t;
typedef __attribute__((ext_vector_type(16))) __bf16 v16bf;
typedef __attribute__((ext_vector_type(8)))  float  v8f;
typedef __attribute__((ext_vector_type(4)))  unsigned int u32x4;
typedef __attribute__((ext_vector_type(8)))  unsigned int u32x8;

// CDNA5 has single-instruction V_TANH_F32 (TRANS32, co-executes with WMMA pipe).
__device__ __forceinline__ float fast_tanh(float x) {
#if __has_builtin(__builtin_amdgcn_tanhf)
  return __builtin_amdgcn_tanhf(x);
#else
  return tanhf(x);
#endif
}
// sigmoid(x) = 0.5*tanh(0.5x)+0.5 : one trans op + one FMA (vs. full div sequence)
__device__ __forceinline__ float sigmoidf_(float x) {
  return __builtin_fmaf(0.5f, fast_tanh(0.5f * x), 0.5f);
}

// ---------------- WMMA fragment helpers (ISA 7.12.2 layouts) ----------------
// A tile 16x32 bf16: lane L holds row (L&15); K = kh..kh+7 and 16+kh..16+kh+7, kh = (L>=16)?8:0
__device__ __forceinline__ v16bf load_a_tile(const bf16_t* base /* at (row0,k0) */, int ldk, int lane) {
  const bf16_t* p = base + (size_t)(lane & 15) * ldk + (((lane >> 4) & 1) << 3);
  v16bf a;
#pragma unroll
  for (int e = 0; e < 8; ++e) a[e] = p[e];
#pragma unroll
  for (int e = 0; e < 8; ++e) a[8 + e] = p[16 + e];
  return a;
}

// B tile 32x16 bf16 taken as W^T: B[k][n] = W[n][k], W row-major [Nout][K].
// lane L holds col (L&15); K = kh..kh+7 and 16+kh..16+kh+7, kh = (L>=16)?8:0
__device__ __forceinline__ v16bf load_b_tile(const bf16_t* W, int n0, int k0, int ldk, int lane, int nmax) {
  int n  = n0 + (lane & 15);
  int kh = ((lane >> 4) & 1) << 3;
  v16bf b;
  if (n < nmax) {
    const bf16_t* p = W + (size_t)n * ldk + k0 + kh;
#pragma unroll
    for (int e = 0; e < 16; ++e) b[e] = p[(e < 8) ? e : 8 + e];
  } else {
#pragma unroll
    for (int e = 0; e < 16; ++e) b[e] = (bf16_t)0.0f;
  }
  return b;
}

__device__ __forceinline__ v8f wmma_bf16(v16bf a, v16bf b, v8f c) {
  return __builtin_amdgcn_wmma_f32_16x16x32_bf16(false, a, false, b, (short)0, c, false, false);
}

// ---------------- Tensor Data Mover: 2-D bf16 tile Global -> LDS ----------------
// D# per CDNA5 ISA. Group0: count=1, lds_addr, 57b global_addr,
// type=2. Group1: data_size=1 (2B), tensor_dim0=cols, tensor_dim1=rows,
// tile_dim0=cols, tile_dim1=rows, tensor_dim0_stride=cols. Groups 2/3 omitted (2-D).
// Caller: one wave issues this, then s_wait_tensorcnt + block barrier.
__device__ __forceinline__ void tdm_load_2d_bf16(const bf16_t* gsrc, bf16_t* lds_dst,
                                                 unsigned rows, unsigned cols) {
  unsigned long long ga = (unsigned long long)(size_t)gsrc;
  unsigned lds = (unsigned)(size_t)lds_dst;   // generic LDS addr: low 32 bits = LDS offset
  u32x4 g0;
  g0[0] = 1u;                                              // count=1 (valid user D#)
  g0[1] = lds;                                             // lds_addr
  g0[2] = (unsigned)(ga & 0xffffffffu);                    // global_addr[31:0]
  g0[3] = (unsigned)((ga >> 32) & 0x01ffffffu) | 0x80000000u; // global_addr[56:32] | type=2
  u32x8 g1;
  g1[0] = 1u << 16;                                        // data_size=1 -> 2 bytes
  g1[1] = (cols & 0xffffu) << 16;                          // tensor_dim0[15:0]
  g1[2] = ((cols >> 16) & 0xffffu) | ((rows & 0xffffu) << 16); // dim0[31:16] | dim1[15:0]
  g1[3] = ((rows >> 16) & 0xffffu) | ((cols & 0xffffu) << 16); // dim1[31:16] | tile_dim0
  g1[4] = rows & 0xffffu;                                  // tile_dim1 (tile_dim2 = 0)
  g1[5] = cols;                                            // tensor_dim0_stride[31:0]
  g1[6] = 0u;                                              // stride hi / dim1_stride
  g1[7] = 0u;
  asm volatile("tensor_load_to_lds %0, %1"
               :
               : "s"(g0), "s"(g1)
               : "memory");
  __builtin_amdgcn_s_wait_tensorcnt(0);
}

// ---------------- conversion / setup kernels ----------------
__global__ void k_f2bf(const float* __restrict__ s, bf16_t* __restrict__ d, int n) {
  int i = blockIdx.x * 256 + threadIdx.x;
  if (i < n) d[i] = (bf16_t)s[i];
}

__global__ void k_f2bf_pad(const float* __restrict__ s, bf16_t* __restrict__ d,
                           int rows, int K, int Kp) {
  int i = blockIdx.x * 256 + threadIdx.x;
  if (i >= rows * Kp) return;
  int r = i / Kp, c = i - r * Kp;
  d[i] = (c < K) ? (bf16_t)s[(size_t)r * K + c] : (bf16_t)0.0f;
}

// gi[t][b][j] = b_ih_enc[j] + sum_k src[t][b][k] * W_ih_enc[j][k]   (K=3, scalar is optimal)
__global__ void k_gi(const float* __restrict__ src, const float* __restrict__ Wih,
                     const float* __restrict__ bih, float* __restrict__ gi) {
  int i = blockIdx.x * 256 + threadIdx.x;
  if (i >= SRCT * BSZ * G3H) return;
  int j  = i % G3H;
  int tb = i / G3H;
  const float* x = src + (size_t)tb * INDIM;
  gi[i] = bih[j] + x[0] * Wih[j * 3] + x[1] * Wih[j * 3 + 1] + x[2] * Wih[j * 3 + 2];
}

// zero output slice 0 (eid logits row 0 and rate row 0)
__global__ void k_zero0(float* __restrict__ out) {
  int i = blockIdx.x * 256 + threadIdx.x;
  if (i < BSZ * EIDN) out[i] = 0.0f;
  if (i < BSZ) out[(size_t)TRGT * BSZ * EIDN + i] = 0.0f;
}

// ---------------- encoder GRU: one persistent block, 32 waves ----------------
__global__ void __launch_bounds__(1024) k_enc_gru(
    const float*  __restrict__ gi_all,   // [SRCT][BSZ][G3H]
    const bf16_t* __restrict__ Whh,      // [G3H][HIDN] bf16
    const float*  __restrict__ bhh,      // [G3H]
    const int*    __restrict__ src_len,  // [BSZ]
    bf16_t*       __restrict__ h_out)    // [BSZ][HIDN]
{
  __shared__ bf16_t hbuf[BSZ * HIDN];    // 64 KB of 320 KB WGP LDS
  const int tid = threadIdx.x, lane = tid & 31, wave = tid >> 5;
  for (int i = tid; i < BSZ * HIDN; i += 1024) hbuf[i] = (bf16_t)0.0f;
  __syncthreads();

  const int N = lane & 15, mhi = ((lane >> 4) & 1) << 3;

  for (int t = 0; t < SRCT; ++t) {
    const float* gi = gi_all + (size_t)t * BSZ * G3H;
    float hnew[4][8];
    int rows[4], cols[4];
#pragma unroll
    for (int q = 0; q < 4; ++q) {
      int tt = wave * 4 + q;           // 128 tile-triples over 32 waves
      int mt = tt >> 5, jt = tt & 31;  // 4 x 32
      int row0 = mt * 16, col0 = jt * 16;
      v8f accR = {}, accZ = {}, accN = {};
      for (int k0 = 0; k0 < HIDN; k0 += 32) {
        v16bf a  = load_a_tile(hbuf + (size_t)row0 * HIDN + k0, HIDN, lane);
        v16bf br = load_b_tile(Whh, col0,            k0, HIDN, lane, G3H);
        v16bf bz = load_b_tile(Whh, col0 + HIDN,     k0, HIDN, lane, G3H);
        v16bf bn = load_b_tile(Whh, col0 + 2 * HIDN, k0, HIDN, lane, G3H);
        accR = wmma_bf16(a, br, accR);
        accZ = wmma_bf16(a, bz, accZ);
        accN = wmma_bf16(a, bn, accN);
      }
      int col = col0 + N;
      rows[q] = row0 + mhi; cols[q] = col;
      float bR = bhh[col], bZ = bhh[col + HIDN], bN = bhh[col + 2 * HIDN];
#pragma unroll
      for (int r = 0; r < 8; ++r) {
        int brow  = row0 + mhi + r;
        float gir = gi[(size_t)brow * G3H + col];
        float giz = gi[(size_t)brow * G3H + col + HIDN];
        float gin = gi[(size_t)brow * G3H + col + 2 * HIDN];
        float rg = sigmoidf_(gir + accR[r] + bR);
        float zg = sigmoidf_(giz + accZ[r] + bZ);
        float ng = fast_tanh(gin + rg * (accN[r] + bN));
        float hold = (float)hbuf[(size_t)brow * HIDN + col];
        float hv = (1.0f - zg) * ng + zg * hold;
        hnew[q][r] = (t < src_len[brow]) ? hv : hold;   // packed-sequence mask
      }
    }
    __syncthreads();                    // all reads of hbuf done
#pragma unroll
    for (int q = 0; q < 4; ++q)
#pragma unroll
      for (int r = 0; r < 8; ++r)
        hbuf[(size_t)(rows[q] + r) * HIDN + cols[q]] = (bf16_t)hnew[q][r];
    __syncthreads();                    // all writes visible before next step
  }
  for (int i = tid; i < BSZ * HIDN; i += 1024) h_out[i] = hbuf[i];
}

// ---------------- decoder step: build x = [emb_dec[eid], rate, pad0] ----------------
__global__ void k_build_x(const int* __restrict__ eid, const float* __restrict__ rate,
                          const float* __restrict__ emb_dec, bf16_t* __restrict__ xbf) {
  int i = blockIdx.x * 256 + threadIdx.x;
  if (i >= BSZ * XPAD) return;
  int b = i / XPAD, c = i - b * XPAD;
  float v = 0.0f;
  if (c < EMB)      v = emb_dec[(size_t)eid[b] * EMB + c];
  else if (c == EMB) v = rate[b];
  xbf[i] = (bf16_t)v;
}

// ---------------- decoder GRUCell: one block, chained WMMA over x then h ----------------
__global__ void __launch_bounds__(1024) k_dec_gru(
    const bf16_t* __restrict__ xbf,  // [BSZ][XPAD]
    const bf16_t* __restrict__ Wih,  // [G3H][XPAD] bf16 (padded)
    const bf16_t* __restrict__ Whh,  // [G3H][HIDN] bf16
    const float*  __restrict__ bih,
    const float*  __restrict__ bhh,
    bf16_t*       __restrict__ hbf)  // [BSZ][HIDN] in/out
{
  __shared__ bf16_t hs[BSZ * HIDN];  // 64 KB
  const int tid = threadIdx.x, lane = tid & 31, wave = tid >> 5;
  if (wave == 0) tdm_load_2d_bf16(hbf, hs, BSZ, HIDN);   // TDM DMA: h -> LDS
  __syncthreads();

  const int N = lane & 15, mhi = ((lane >> 4) & 1) << 3;
  float hnew[4][8]; int rows[4], cols[4];
#pragma unroll
  for (int q = 0; q < 4; ++q) {
    int tt = wave * 4 + q;
    int mt = tt >> 5, jt = tt & 31;
    int row0 = mt * 16, col0 = jt * 16;
    v8f accR = {}, accZ = {}, accNi = {}, accNh = {};
    // gi part: K = 288 over padded x (r and z chain into the same accumulators)
    for (int k0 = 0; k0 < XPAD; k0 += 32) {
      v16bf a = load_a_tile(xbf + (size_t)row0 * XPAD + k0, XPAD, lane);
      accR  = wmma_bf16(a, load_b_tile(Wih, col0,            k0, XPAD, lane, G3H), accR);
      accZ  = wmma_bf16(a, load_b_tile(Wih, col0 + HIDN,     k0, XPAD, lane, G3H), accZ);
      accNi = wmma_bf16(a, load_b_tile(Wih, col0 + 2 * HIDN, k0, XPAD, lane, G3H), accNi);
    }
    // gh part: K = 512 over h (n-gate kept separate for r * h_n)
    for (int k0 = 0; k0 < HIDN; k0 += 32) {
      v16bf a = load_a_tile(hs + (size_t)row0 * HIDN + k0, HIDN, lane);
      accR  = wmma_bf16(a, load_b_tile(Whh, col0,            k0, HIDN, lane, G3H), accR);
      accZ  = wmma_bf16(a, load_b_tile(Whh, col0 + HIDN,     k0, HIDN, lane, G3H), accZ);
      accNh = wmma_bf16(a, load_b_tile(Whh, col0 + 2 * HIDN, k0, HIDN, lane, G3H), accNh);
    }
    int col = col0 + N;
    rows[q] = row0 + mhi; cols[q] = col;
    float bR  = bih[col] + bhh[col];
    float bZ  = bih[col + HIDN] + bhh[col + HIDN];
    float bNi = bih[col + 2 * HIDN], bNh = bhh[col + 2 * HIDN];
#pragma unroll
    for (int r = 0; r < 8; ++r) {
      float rg = sigmoidf_(accR[r] + bR);
      float zg = sigmoidf_(accZ[r] + bZ);
      float ng = fast_tanh(accNi[r] + bNi + rg * (accNh[r] + bNh));
      float hold = (float)hs[(size_t)(rows[q] + r) * HIDN + col];
      hnew[q][r] = (1.0f - zg) * ng + zg * hold;
    }
  }
#pragma unroll
  for (int q = 0; q < 4; ++q)
#pragma unroll
    for (int r = 0; r < 8; ++r)
      hbf[(size_t)(rows[q] + r) * HIDN + cols[q]] = (bf16_t)hnew[q][r];
}

// ---------------- vocabulary projection: logits = h @ W_eid^T + b ----------------
__global__ void __launch_bounds__(256) k_logits(
    const bf16_t* __restrict__ hbf, const bf16_t* __restrict__ Weid,
    const float* __restrict__ beid, float* __restrict__ logits)
{
  __shared__ bf16_t hs[BSZ * HIDN];    // 64 KB staged h (L2-resident source)
  const int tid = threadIdx.x, lane = tid & 31, wave = tid >> 5;
  if (wave == 0) tdm_load_2d_bf16(hbf, hs, BSZ, HIDN);   // TDM DMA: h -> LDS
  __syncthreads();

  const int n0 = blockIdx.x * 64;      // 64 vocab cols per block
  const int N = lane & 15, mhi = ((lane >> 4) & 1) << 3;
  for (int tl = wave * 2; tl < wave * 2 + 2; ++tl) {   // 16 tiles / 8 waves
    int mt = tl & 3, nt = tl >> 2;
    int row0 = mt * 16, col0 = n0 + nt * 16;
    v8f acc = {};
    for (int k0 = 0; k0 < HIDN; k0 += 32) {
      if (k0 + 32 < HIDN) {            // stream next K-chunk of weights from L2
        int n = col0 + N;
        if (n < EIDN) __builtin_prefetch(Weid + (size_t)n * HIDN + k0 + 32, 0, 1);
      }
      v16bf a = load_a_tile(hs + (size_t)row0 * HIDN + k0, HIDN, lane);
      v16bf b = load_b_tile(Weid, col0, k0, HIDN, lane, EIDN);
      acc = wmma_bf16(a, b, acc);
    }
    int col = col0 + N;
    if (col < EIDN) {
      float bias = beid[col];
#pragma unroll
      for (int r = 0; r < 8; ++r)
        logits[(size_t)(row0 + mhi + r) * EIDN + col] = acc[r] + bias;
    }
  }
}

// ---------------- per-row max/argmax + log-sum-exp ----------------
__global__ void __launch_bounds__(256) k_rowred(const float* __restrict__ logits,
                                                float* __restrict__ lse, int* __restrict__ amax) {
  __shared__ float sv[256]; __shared__ int si[256]; __shared__ float ss[256];
  const int b = blockIdx.x, tid = threadIdx.x;
  const float* row = logits + (size_t)b * EIDN;
  float m = -3.4e38f; int mi = 0;
  for (int e = tid; e < EIDN; e += 256) { float v = row[e]; if (v > m) { m = v; mi = e; } }
  sv[tid] = m; si[tid] = mi;
  __syncthreads();
  for (int s = 128; s > 0; s >>= 1) {
    if (tid < s && sv[tid + s] > sv[tid]) { sv[tid] = sv[tid + s]; si[tid] = si[tid + s]; }
    __syncthreads();
  }
  float rmax = sv[0];
  float acc = 0.0f;
  for (int e = tid; e < EIDN; e += 256) acc += __expf(row[e] - rmax);
  ss[tid] = acc;
  __syncthreads();
  for (int s = 128; s > 0; s >>= 1) {
    if (tid < s) ss[tid] += ss[tid + s];
    __syncthreads();
  }
  if (tid == 0) { lse[b] = rmax + __logf(ss[0]); amax[b] = si[0]; }
}

// log_softmax written straight into the output slice
__global__ void k_norm(const float* __restrict__ logits, const float* __restrict__ lse,
                       float* __restrict__ outslice) {
  int i = blockIdx.x * 256 + threadIdx.x;
  if (i < BSZ * EIDN) { int b = i / EIDN; outslice[i] = logits[i] - lse[b]; }
}

// ---------------- rate head: relu(concat(emb_mt[idx], h) @ W_r1^T + b) @ W_r2^T ----------------
__global__ void __launch_bounds__(1024) k_rate(
    const bf16_t* __restrict__ hbf, const int* __restrict__ idx,
    const float* __restrict__ emb_mt, const bf16_t* __restrict__ Wr1,
    const float* __restrict__ br1, const float* __restrict__ Wr2,
    const float* __restrict__ br2, bf16_t* __restrict__ r1ws,
    float* __restrict__ rate_ws, float* __restrict__ out_rate)
{
  __shared__ bf16_t ys[BSZ * YK];   // 96 KB (CDNA5 WGP LDS is 320 KB)
  const int tid = threadIdx.x, lane = tid & 31, wave = tid >> 5;
  for (int i = tid; i < BSZ * YK; i += 1024) {
    int b = i / YK, c = i - b * YK;
    float v = (c < EMB) ? emb_mt[(size_t)idx[b] * EMB + c]
                        : (float)hbf[(size_t)b * HIDN + (c - EMB)];
    ys[i] = (bf16_t)v;
  }
  __syncthreads();
  const int N = lane & 15, mhi = ((lane >> 4) & 1) << 3;
#pragma unroll
  for (int q = 0; q < 4; ++q) {
    int tt = wave * 4 + q;
    int mt = tt >> 5, jt = tt & 31;
    int row0 = mt * 16, col0 = jt * 16;
    v8f acc = {};
    for (int k0 = 0; k0 < YK; k0 += 32) {
      v16bf a = load_a_tile(ys + (size_t)row0 * YK + k0, YK, lane);
      v16bf b = load_b_tile(Wr1, col0, k0, YK, lane, HIDN);
      acc = wmma_bf16(a, b, acc);
    }
    int col = col0 + N;
    float bias = br1[col];
#pragma unroll
    for (int r = 0; r < 8; ++r) {
      float v = acc[r] + bias;
      r1ws[(size_t)(row0 + mhi + r) * HIDN + col] = (bf16_t)(v > 0.0f ? v : 0.0f);
    }
  }
  __syncthreads();   // workgroup fence: r1ws writes visible to the dot below
  if (tid < BSZ) {
    float s = br2[0];
    for (int j = 0; j < HIDN; ++j) s += (float)r1ws[(size_t)tid * HIDN + j] * Wr2[j];
    float rt = sigmoidf_(s);
    rate_ws[tid] = rt;
    out_rate[tid] = rt;
  }
}

// ---------------- host orchestration ----------------
extern "C" void kernel_launch(void* const* d_in, const int* in_sizes, int n_in,
                              void* d_out, int out_size, void* d_ws, size_t ws_size,
                              hipStream_t stream) {
  (void)in_sizes; (void)n_in; (void)out_size; (void)ws_size;
  const float* src      = (const float*)d_in[0];
  const int*   src_len  = (const int*)  d_in[1];
  const int*   trg_eid  = (const int*)  d_in[2];
  const float* trg_rate = (const float*)d_in[3];
  const float* W_ih_enc = (const float*)d_in[4];
  const float* W_hh_enc = (const float*)d_in[5];
  const float* b_ih_enc = (const float*)d_in[6];
  const float* b_hh_enc = (const float*)d_in[7];
  const float* emb_dec  = (const float*)d_in[8];
  const float* W_ih_dec = (const float*)d_in[9];
  const float* W_hh_dec = (const float*)d_in[10];
  const float* b_ih_dec = (const float*)d_in[11];
  const float* b_hh_dec = (const float*)d_in[12];
  const float* emb_mt   = (const float*)d_in[13];
  const float* W_eid    = (const float*)d_in[14];
  const float* b_eid    = (const float*)d_in[15];
  const float* W_r1     = (const float*)d_in[16];
  const float* b_r1     = (const float*)d_in[17];
  const float* W_r2     = (const float*)d_in[18];
  const float* b_r2     = (const float*)d_in[19];
  float* out = (float*)d_out;

  char* ws = (char*)d_ws;
  size_t off = 0;
  auto alloc = [&](size_t bytes) -> void* {
    void* p = ws + off;
    off = (off + bytes + 255) & ~(size_t)255;
    return p;
  };
  float*  giW     = (float*) alloc((size_t)SRCT * BSZ * G3H * 4);
  bf16_t* WhhEnc  = (bf16_t*)alloc((size_t)G3H * HIDN * 2);
  bf16_t* WihDec  = (bf16_t*)alloc((size_t)G3H * XPAD * 2);
  bf16_t* WhhDec  = (bf16_t*)alloc((size_t)G3H * HIDN * 2);
  bf16_t* WeidBf  = (bf16_t*)alloc((size_t)EIDN * HIDN * 2);
  bf16_t* Wr1Bf   = (bf16_t*)alloc((size_t)HIDN * YK * 2);
  bf16_t* hbf     = (bf16_t*)alloc((size_t)BSZ * HIDN * 2);
  bf16_t* xbf     = (bf16_t*)alloc((size_t)BSZ * XPAD * 2);
  bf16_t* r1ws    = (bf16_t*)alloc((size_t)BSZ * HIDN * 2);
  float*  logits  = (float*) alloc((size_t)BSZ * EIDN * 4);
  float*  lse     = (float*) alloc((size_t)BSZ * 4);
  int*    idxb    = (int*)   alloc((size_t)BSZ * 4);
  float*  ratews  = (float*) alloc((size_t)BSZ * 4);

  // weight conversions fp32 -> bf16 (done every call; deterministic)
  k_f2bf    <<<(G3H * HIDN + 255) / 256, 256, 0, stream>>>(W_hh_enc, WhhEnc, G3H * HIDN);
  k_f2bf_pad<<<(G3H * XPAD + 255) / 256, 256, 0, stream>>>(W_ih_dec, WihDec, G3H, EMB + 1, XPAD);
  k_f2bf    <<<(G3H * HIDN + 255) / 256, 256, 0, stream>>>(W_hh_dec, WhhDec, G3H * HIDN);
  k_f2bf    <<<(EIDN * HIDN + 255) / 256, 256, 0, stream>>>(W_eid, WeidBf, EIDN * HIDN);
  k_f2bf    <<<(HIDN * YK + 255) / 256, 256, 0, stream>>>(W_r1, Wr1Bf, HIDN * YK);

  // encoder input projection (K=3) and output slice-0 zeroing
  k_gi   <<<(SRCT * BSZ * G3H + 255) / 256, 256, 0, stream>>>(src, W_ih_enc, b_ih_enc, giW);
  k_zero0<<<(BSZ * EIDN + 255) / 256, 256, 0, stream>>>(out);

  // encoder GRU (sequential over 128 steps inside one persistent block)
  k_enc_gru<<<1, 1024, 0, stream>>>(giW, WhhEnc, b_hh_enc, src_len, hbf);

  // autoregressive decoder: 49 steps
  for (int s = 0; s < TRGT - 1; ++s) {
    const int*   eprev = (s == 0) ? trg_eid  : idxb;    // trg_eid row 0 == first 64 ints
    const float* rprev = (s == 0) ? trg_rate : ratews;  // trg_rate row 0
    k_build_x<<<(BSZ * XPAD + 255) / 256, 256, 0, stream>>>(eprev, rprev, emb_dec, xbf);
    k_dec_gru<<<1, 1024, 0, stream>>>(xbf, WihDec, WhhDec, b_ih_dec, b_hh_dec, hbf);
    k_logits <<<(EIDN + 63) / 64, 256, 0, stream>>>(hbf, WeidBf, b_eid, logits);
    k_rowred <<<BSZ, 256, 0, stream>>>(logits, lse, idxb);
    k_norm   <<<(BSZ * EIDN + 255) / 256, 256, 0, stream>>>(
        logits, lse, out + (size_t)(s + 1) * BSZ * EIDN);
    k_rate   <<<1, 1024, 0, stream>>>(hbf, idxb, emb_mt, Wr1Bf, b_r1, W_r2, b_r2,
                                      r1ws, ratews,
                                      out + (size_t)TRGT * BSZ * EIDN + (size_t)(s + 1) * BSZ);
  }
}
